// AnchorDeformAtt_14018773254655
// MI455X (gfx1250) — matrix operand order repaired and checked by reading, not compiled, not run on hardware
//
#include <hip/hip_runtime.h>
#include <hip/hip_bf16.h>

// ---------------------------------------------------------------------------
// AnchorDeformAtt for MI455X (gfx1250, wave32, WMMA + TDM)
// ---------------------------------------------------------------------------

typedef __attribute__((ext_vector_type(16))) __bf16 v16bf;
typedef __attribute__((ext_vector_type(8)))  float  v8f;

#define B_DIM  2
#define C_DIM  256
#define L_DIM  4096
#define W_DIM  64
#define NH     8
#define NP     16
#define HD     32          // C / NH
#define OCAT   656         // 256 (mem) + 16 (size) + 256 (offset) + 128 (attn)
#define OPAD   704         // OCAT padded to 64-row multiple
#define KSTR   40          // padded bf16 K-stride for B tiles (80B rows)
#define ASTR   264         // padded bf16 K-stride for A panel (528B rows)

// P row layout (per position): [0,256) memory, [256,272) size logits,
// [272,528) offset logits, [528,656) attn logits.

#if defined(__AMDGCN__) && __has_builtin(__builtin_amdgcn_tensor_load_to_lds)
#define USE_TDM 1
#else
#define USE_TDM 0
#endif

#if USE_TDM
typedef __attribute__((ext_vector_type(4))) unsigned tdm_u4;
typedef __attribute__((ext_vector_type(8))) int      tdm_i8;
typedef __attribute__((ext_vector_type(4))) int      tdm_i4;

// Load a 64-row x 256-col bf16 tile (row stride 256 elements in memory) into
// LDS with 16B padding appended per 512B row (-> ASTR=264 bf16 LDS stride).
// D# per CDNA5 ISA 8.3-8.6: group0 = {count=1 | lds_addr | global_addr | type=2},
// group1 = {flags/pad | tensor_dim0 | tensor_dim1 | tile_dim0 | tile_dim1 |
//           tensor_dim0_stride}.
__device__ __forceinline__ void tdm_load_panel(const __bf16* gsrc, void* lds_dst,
                                               unsigned tensor_dim1) {
    unsigned long long ga = (unsigned long long)(uintptr_t)gsrc;
    unsigned lds_off = (unsigned)(uintptr_t)lds_dst;
    tdm_u4 g0 = { 1u,                                   // count=1 (valid D#)
                  lds_off,                              // lds_addr  [63:32]
                  (unsigned)(ga & 0xFFFFFFFFu),         // global_addr[95:64]
                  (unsigned)(((ga >> 32) & 0x1FFFFFFull) // global_addr[120:96]
                             | 0x80000000u) };          // type=2    [127:126]
    tdm_i8 g1 = { (int)((1u << 16)    // data_size = 2 bytes
                      | (1u << 20)    // pad_enable
                      | (6u << 22)    // pad_interval: every 512B
                      | (3u << 25)),  // pad_amount:   16B (4 dwords)
                  (int)(256u << 16),          // tensor_dim0 = 256  [79:48]
                  (int)(tensor_dim1 << 16),   // tensor_dim1        [111:80]
                  (int)(256u << 16),          // tile_dim0 = 256    [127:112]
                  64,                         // tile_dim1 = 64     [143:128]
                  256,                        // tensor_dim0_stride [207:160]
                  0, 0 };
    tdm_i4 z4 = { 0, 0, 0, 0 };
#if __clang_major__ >= 23
    tdm_i8 z8 = { 0, 0, 0, 0, 0, 0, 0, 0 };
    __builtin_amdgcn_tensor_load_to_lds(g0, g1, z4, z4, z8, 0);
#else
    __builtin_amdgcn_tensor_load_to_lds(g0, g1, z4, z4, 0);
#endif
#if __has_builtin(__builtin_amdgcn_s_wait_tensorcnt)
    __builtin_amdgcn_s_wait_tensorcnt(0);
#else
    asm volatile("s_wait_tensorcnt 0x0" ::: "memory");
#endif
}
#endif  // USE_TDM

// Fallback / host-pass path: manual vectorized panel copy (64 x 256 bf16).
__device__ __forceinline__ void manual_load_panel(const __bf16* gsrc,
                                                  __bf16* sA16, int t) {
    const int m = t >> 2, q = t & 3;
    const uint4* src =
        reinterpret_cast<const uint4*>(gsrc + (size_t)m * C_DIM + q * 64);
    uint4* dst = reinterpret_cast<uint4*>(&sA16[m * ASTR + q * 64]);
#pragma unroll
    for (int j = 0; j < 8; ++j) dst[j] = src[j];
}

__device__ __forceinline__ float sigmoidf_(float x) {
    return 1.0f / (1.0f + __expf(-x));
}

// ---------------------------------------------------------------------------
// Kernel 0: pack weights (bf16) + biases
// ---------------------------------------------------------------------------
__global__ __launch_bounds__(256)
void prep_kernel(const float* __restrict__ vw, const float* __restrict__ vb,
                 const float* __restrict__ sw, const float* __restrict__ sb,
                 const float* __restrict__ aw, const float* __restrict__ ab,
                 const float* __restrict__ tw, const float* __restrict__ tb,
                 const float* __restrict__ ow,
                 __bf16* __restrict__ Wcat, float* __restrict__ bcat,
                 __bf16* __restrict__ Wout) {
    int idx = blockIdx.x * 256 + threadIdx.x;
    if (idx < OPAD * C_DIM) {
        int row = idx >> 8, col = idx & 255;
        float v = 0.0f;
        if (row < 256)      v = vw[row * 256 + col];
        else if (row < 272) v = sw[(row - 256) * 256 + col];
        else if (row < 528) v = aw[(row - 272) * 256 + col];
        else if (row < 656) v = tw[(row - 528) * 256 + col];
        Wcat[idx] = (__bf16)v;
    }
    if (idx < 256 * 256) Wout[idx] = (__bf16)ow[idx];
    if (idx < OPAD) {
        float v = 0.0f;
        if (idx < 256)      v = vb[idx];
        else if (idx < 272) v = sb[idx - 256];
        else if (idx < 528) v = ab[idx - 272];
        else if (idx < 656) v = tb[idx - 528];
        bcat[idx] = v;
    }
}

// ---------------------------------------------------------------------------
// Kernel 1: fused projection GEMM  P[b][l][o] = sum_c Wcat[o][c]*feat[b][c][l]+b
// Block: 256 thr (8 waves), tile M=64 x N=128, K step 32, bf16 WMMA.
// A panel (64x256) resident in LDS (TDM-loaded); aliases epilogue staging.
// ---------------------------------------------------------------------------
__global__ __launch_bounds__(256)
void fused_proj_kernel(const float* __restrict__ feat,
                       const __bf16* __restrict__ Wcat,
                       const float* __restrict__ bcat,
                       float* __restrict__ P) {
    // sU: A panel (64*264*2 = 33792B) during mainloop; sC (128*68*4 = 34816B)
    // during epilogue (A is dead by then).
    __shared__ __align__(16) unsigned char sU[128 * 68 * 4];
    __shared__ __align__(16) __bf16 sB[128 * KSTR];
    __bf16* sA16 = reinterpret_cast<__bf16*>(sU);
    float*  sC   = reinterpret_cast<float*>(sU);

    const int bid = blockIdx.x;
    const int mB  = bid >> 6;            // 0..10
    const int nb  = bid & 63;
    const int b   = nb >> 5;             // batch
    const int l0  = (nb & 31) << 7;      // column base within batch (128-wide)
    const int t    = threadIdx.x;
    const int wave = t >> 5;
    const int lane = t & 31;
    const int wave_m = wave & 3;         // 4 strips of 16 rows
    const int wave_n = wave >> 2;        // 2 strips of 64 cols

    v8f acc[4];
#pragma unroll
    for (int s = 0; s < 4; ++s)
#pragma unroll
        for (int r = 0; r < 8; ++r) acc[s][r] = 0.0f;

    const float* featB = feat + (size_t)b * C_DIM * L_DIM;

    // ---- A panel: 64 x 256 bf16, loaded once (TDM if available) ----
#if USE_TDM
    if (wave == 0)
        tdm_load_panel(Wcat + (size_t)(mB * 64) * C_DIM, sA16, OPAD);
#else
    manual_load_panel(Wcat + (size_t)(mB * 64) * C_DIM, sA16, t);
#endif
    __syncthreads();

    // B-fill mapping: thread owns k-pair kp (k = 2*kp, 2*kp+1) x 8 positions.
    const int kp = t & 15;
    const int n8 = t >> 4;               // 0..15 -> positions n8*8..n8*8+7

    for (int k0 = 0; k0 < C_DIM; k0 += 32) {
        // ---- B tile: transpose feat (channel-major) into sB[n][k] ----
        {
            const float* r0 = featB + (size_t)(k0 + 2 * kp)     * L_DIM + l0 + n8 * 8;
            const float* r1 = featB + (size_t)(k0 + 2 * kp + 1) * L_DIM + l0 + n8 * 8;
            const float4 a0 = reinterpret_cast<const float4*>(r0)[0];
            const float4 a1 = reinterpret_cast<const float4*>(r0)[1];
            const float4 b0 = reinterpret_cast<const float4*>(r1)[0];
            const float4 b1 = reinterpret_cast<const float4*>(r1)[1];
            const float e0[8] = {a0.x, a0.y, a0.z, a0.w, a1.x, a1.y, a1.z, a1.w};
            const float e1[8] = {b0.x, b0.y, b0.z, b0.w, b1.x, b1.y, b1.z, b1.w};
#pragma unroll
            for (int i = 0; i < 8; ++i) {
                union { __bf16 h[2]; unsigned u; } p;
                p.h[0] = (__bf16)e0[i];
                p.h[1] = (__bf16)e1[i];
                *reinterpret_cast<unsigned*>(&sB[(n8 * 8 + i) * KSTR + 2 * kp]) = p.u;
            }
            if (k0 + 32 < C_DIM) {
                __builtin_prefetch(featB + (size_t)(k0 + 32 + 2 * kp) * L_DIM +
                                       l0 + n8 * 8, 0, 1);
            }
        }
        __syncthreads();

        // A fragment from resident panel: lane (m = lane%16, grp = lane/16):
        //   elems 0..15 -> K = k0+grp*16 .. k0+grp*16+15
        const int row = wave_m * 16 + (lane & 15);
        const int grp = lane >> 4;
        union { v16bf v; uint4 q[2]; } af;
        af.q[0] = *reinterpret_cast<const uint4*>(&sA16[row * ASTR + k0 + grp * 16]);
        af.q[1] = *reinterpret_cast<const uint4*>(&sA16[row * ASTR + k0 + grp * 16 + 8]);

#pragma unroll
        for (int sub = 0; sub < 4; ++sub) {
            // B fragment: lane (n = lane%16, grp): K = grp*16 .. grp*16+15
            const int coln = wave_n * 64 + sub * 16 + (lane & 15);
            union { v16bf v; uint4 q[2]; } bfr;
            const uint4* bp =
                reinterpret_cast<const uint4*>(&sB[coln * KSTR + grp * 16]);
            bfr.q[0] = bp[0];
            bfr.q[1] = bp[1];
            acc[sub] = __builtin_amdgcn_wmma_f32_16x16x32_bf16(
                false, af.v, false, bfr.v, (short)0, acc[sub], false, false);
        }
        __syncthreads();
    }

    // Epilogue: transpose through LDS (reusing A panel space), emit
    // position-major rows of P.
#pragma unroll
    for (int sub = 0; sub < 4; ++sub) {
        const int coln  = wave_n * 64 + sub * 16 + (lane & 15);
        const int mbase = wave_m * 16 + (lane >> 4) * 8;
#pragma unroll
        for (int r = 0; r < 8; ++r)
            sC[coln * 68 + mbase + r] = acc[sub][r];
    }
    __syncthreads();

    const int pos  = t >> 1;
    const int half = t & 1;
    const int oBase = mB * 64 + half * 32;
    const size_t outBase = (size_t)(b * L_DIM + l0 + pos) * OCAT;
#pragma unroll
    for (int i = 0; i < 32; ++i) {
        int o = oBase + i;
        if (o < OCAT)
            P[outBase + o] = sC[pos * 68 + half * 32 + i] + bcat[o];
    }
}

// ---------------------------------------------------------------------------
// Kernel 2: deformable sampling + softmax.  One wave per (b, head, position).
// Lanes p=lane%16 own sampling point p; all 32 lanes own one head channel.
// Gathers are 32 contiguous floats (128 B) from P's memory section.
// ---------------------------------------------------------------------------
__global__ __launch_bounds__(256)
void sample_kernel(const float* __restrict__ P, float* __restrict__ OA) {
    const int t    = threadIdx.x;
    const int wave = t >> 5;
    const int lane = t & 31;
    const int wg   = blockIdx.x * 8 + wave;      // 0 .. 65535
    const int b    = wg >> 15;
    const int rem  = wg & 32767;
    const int h    = rem >> 12;
    const int l    = rem & 4095;

    const float* Pp = P + (size_t)(b * L_DIM + l) * OCAT;
    const int p = lane & 15;

    const float ox   = sigmoidf_(Pp[272 + h * 32 + p * 2]);
    const float oy   = sigmoidf_(Pp[272 + h * 32 + p * 2 + 1]);
    const float alog = Pp[528 + h * 16 + p];
    const float sx = fminf(fmaxf(sigmoidf_(Pp[256 + h * 2]),     0.25f), 0.75f);
    const float sy = fminf(fmaxf(sigmoidf_(Pp[256 + h * 2 + 1]), 0.25f), 0.75f);

    const float inv = 1.0f / (64.0f + 1e-6f);
    const float cx = ((float)(l & 63) + 0.5f) * inv;
    const float cy = ((float)(l >> 6) + 0.5f) * inv;

    const float gx = fminf(fmaxf(cx - 0.5f * sx + ox * sx, 0.0f), 1.0f);
    const float gy = fminf(fmaxf(cy - 0.5f * sy + oy * sy, 0.0f), 1.0f);
    const float ix = gx * 63.0f, iy = gy * 63.0f;
    const float fx0 = floorf(ix), fy0 = floorf(iy);
    const float wx = ix - fx0, wy = iy - fy0;
    const int x0 = min((int)fx0, 63), x1 = min(x0 + 1, 63);
    const int y0 = min((int)fy0, 63), y1 = min(y0 + 1, 63);

    // softmax across the 16 points (values duplicated in both half-waves)
    float mx = alog;
#pragma unroll
    for (int o = 1; o < 16; o <<= 1) mx = fmaxf(mx, __shfl_xor(mx, o, 32));
    const float e = __expf(alog - mx);
    float ssum = e;
#pragma unroll
    for (int o = 1; o < 16; o <<= 1) ssum += __shfl_xor(ssum, o, 32);
    const float attw = e / ssum;

    const float w00 = (1.0f - wx) * (1.0f - wy) * attw;
    const float w01 = wx * (1.0f - wy) * attw;
    const float w10 = (1.0f - wx) * wy * attw;
    const float w11 = wx * wy * attw;
    const int i00 = y0 * W_DIM + x0, i01 = y0 * W_DIM + x1;
    const int i10 = y1 * W_DIM + x0, i11 = y1 * W_DIM + x1;

    const float* Vb = P + (size_t)b * L_DIM * OCAT + h * HD + lane;
    float acc = 0.0f;
#pragma unroll
    for (int q = 0; q < 16; ++q) {
        const float a00 = __shfl(w00, q, 32), a01 = __shfl(w01, q, 32);
        const float a10 = __shfl(w10, q, 32), a11 = __shfl(w11, q, 32);
        const int j00 = __shfl(i00, q, 32), j01 = __shfl(i01, q, 32);
        const int j10 = __shfl(i10, q, 32), j11 = __shfl(i11, q, 32);
        acc += a00 * Vb[(size_t)j00 * OCAT] + a01 * Vb[(size_t)j01 * OCAT]
             + a10 * Vb[(size_t)j10 * OCAT] + a11 * Vb[(size_t)j11 * OCAT];
    }
    OA[(size_t)(b * L_DIM + l) * C_DIM + h * HD + lane] = acc;
}

// ---------------------------------------------------------------------------
// Kernel 3: out_proj GEMM + BN affine, coalesced (B,C,H,W) store.
// ---------------------------------------------------------------------------
__global__ __launch_bounds__(256)
void out_proj_kernel(const float* __restrict__ OA,
                     const __bf16* __restrict__ Wout,
                     const float* __restrict__ gamma,
                     const float* __restrict__ beta,
                     float* __restrict__ out) {
    __shared__ __align__(16) __bf16 sA16[64 * ASTR];
    __shared__ __align__(16) __bf16 sB[128 * KSTR];

    const int bid = blockIdx.x;
    const int mB  = bid >> 6;            // 0..3
    const int nb  = bid & 63;
    const int b   = nb >> 5;
    const int l0  = (nb & 31) << 7;
    const int t    = threadIdx.x;
    const int wave = t >> 5;
    const int lane = t & 31;
    const int wave_m = wave & 3;
    const int wave_n = wave >> 2;

    v8f acc[4];
#pragma unroll
    for (int s = 0; s < 4; ++s)
#pragma unroll
        for (int r = 0; r < 8; ++r) acc[s][r] = 0.0f;

    // ---- A panel: 64 x 256 bf16, loaded once (TDM if available) ----
#if USE_TDM
    if (wave == 0)
        tdm_load_panel(Wout + (size_t)(mB * 64) * C_DIM, sA16, C_DIM);
#else
    manual_load_panel(Wout + (size_t)(mB * 64) * C_DIM, sA16, t);
#endif
    __syncthreads();

    for (int k0 = 0; k0 < C_DIM; k0 += 32) {
        {
            // OA is position-major: thread's 16 values are contiguous in k.
            const int n = t >> 1, half = t & 1;
            const float* src =
                OA + (size_t)(b * L_DIM + l0 + n) * C_DIM + k0 + half * 16;
            const float4 f0 = reinterpret_cast<const float4*>(src)[0];
            const float4 f1 = reinterpret_cast<const float4*>(src)[1];
            const float4 f2 = reinterpret_cast<const float4*>(src)[2];
            const float4 f3 = reinterpret_cast<const float4*>(src)[3];
            const float e[16] = {f0.x, f0.y, f0.z, f0.w, f1.x, f1.y, f1.z, f1.w,
                                 f2.x, f2.y, f2.z, f2.w, f3.x, f3.y, f3.z, f3.w};
            union { __bf16 h[16]; uint4 q[2]; } pk;
#pragma unroll
            for (int i = 0; i < 16; ++i) pk.h[i] = (__bf16)e[i];
            uint4* dst = reinterpret_cast<uint4*>(&sB[n * KSTR + half * 16]);
            dst[0] = pk.q[0];
            dst[1] = pk.q[1];
            if (k0 + 32 < C_DIM)
                __builtin_prefetch(src + 32, 0, 1);
        }
        __syncthreads();

        const int row = wave_m * 16 + (lane & 15);
        const int grp = lane >> 4;
        union { v16bf v; uint4 q[2]; } af;
        af.q[0] = *reinterpret_cast<const uint4*>(&sA16[row * ASTR + k0 + grp * 16]);
        af.q[1] = *reinterpret_cast<const uint4*>(&sA16[row * ASTR + k0 + grp * 16 + 8]);

#pragma unroll
        for (int sub = 0; sub < 4; ++sub) {
            const int coln = wave_n * 64 + sub * 16 + (lane & 15);
            union { v16bf v; uint4 q[2]; } bfr;
            const uint4* bp =
                reinterpret_cast<const uint4*>(&sB[coln * KSTR + grp * 16]);
            bfr.q[0] = bp[0];
            bfr.q[1] = bp[1];
            acc[sub] = __builtin_amdgcn_wmma_f32_16x16x32_bf16(
                false, af.v, false, bfr.v, (short)0, acc[sub], false, false);
        }
        __syncthreads();
    }

    // Fused BN epilogue, channel-major store (lane-contiguous along L).
    const float rs = rsqrtf(1.0f + 1e-5f);
#pragma unroll
    for (int sub = 0; sub < 4; ++sub) {
        const int col = l0 + wave_n * 64 + sub * 16 + (lane & 15);
#pragma unroll
        for (int r = 0; r < 8; ++r) {
            const int m  = mB * 64 + wave_m * 16 + (lane >> 4) * 8 + r;
            const float sc = gamma[m] * rs;
            out[(size_t)b * C_DIM * L_DIM + (size_t)m * L_DIM + col] =
                acc[sub][r] * sc + beta[m];
        }
    }
}

// ---------------------------------------------------------------------------
// Launcher
// ---------------------------------------------------------------------------
extern "C" void kernel_launch(void* const* d_in, const int* in_sizes, int n_in,
                              void* d_out, int out_size, void* d_ws, size_t ws_size,
                              hipStream_t stream) {
    const float* feat = (const float*)d_in[0];
    const float* vw   = (const float*)d_in[1];
    const float* vb   = (const float*)d_in[2];
    const float* sw   = (const float*)d_in[3];
    const float* sb   = (const float*)d_in[4];
    const float* aw   = (const float*)d_in[5];
    const float* ab   = (const float*)d_in[6];
    const float* tw   = (const float*)d_in[7];
    const float* tb   = (const float*)d_in[8];
    const float* ow   = (const float*)d_in[9];
    const float* gamma= (const float*)d_in[10];
    const float* beta = (const float*)d_in[11];

    char* ws = (char*)d_ws;
    // Workspace layout (all 16B aligned):
    __bf16* Wcat = (__bf16*)(ws);                       // 704*256*2   = 360448
    float*  bcat = (float*)(ws + 360448);               // 704*4       = 2816
    __bf16* Wout = (__bf16*)(ws + 363264);              // 256*256*2   = 131072
    float*  P    = (float*)(ws + 494336);               // 2*4096*656*4= 21495808
    float*  OA   = (float*)(ws + 21990144);             // 2*4096*256*4= 8388608
    // total: 30,378,752 bytes

    prep_kernel<<<OPAD, 256, 0, stream>>>(vw, vb, sw, sb, aw, ab, tw, tb, ow,
                                          Wcat, bcat, Wout);
    fused_proj_kernel<<<(OPAD / 64) * 64, 256, 0, stream>>>(feat, Wcat, bcat, P);
    sample_kernel<<<(B_DIM * NH * L_DIM) / 8, 256, 0, stream>>>(P, OA);
    out_proj_kernel<<<(C_DIM / 64) * 64, 256, 0, stream>>>(OA, Wout, gamma, beta,
                                                           (float*)d_out);
}
